// LSM_87625922773089
// MI455X (gfx1250) — matrix operand order repaired and checked by reading, not compiled, not run on hardware
//
#include <hip/hip_runtime.h>
#include <cstdint>

// ---------------------------------------------------------------------------
// Types for WMMA fragments (gfx1250, wave32)
// ---------------------------------------------------------------------------
typedef __attribute__((ext_vector_type(16))) _Float16 v16h;
typedef __attribute__((ext_vector_type(8)))  _Float16 v8h;
typedef __attribute__((ext_vector_type(4)))  _Float16 v4h;
typedef __attribute__((ext_vector_type(8)))  float    v8f;
typedef __attribute__((ext_vector_type(4)))  float    v4f;

union Frag16 { v16h v; v8h h[2]; };

#define EPSV 1e-6f

// ---------------------------------------------------------------------------
// Wave helpers (wave32)
// ---------------------------------------------------------------------------
__device__ __forceinline__ float waveAllMax(float v) {
#pragma unroll
  for (int o = 1; o < 32; o <<= 1) v = fmaxf(v, __shfl_xor(v, o, 32));
  return v;
}
__device__ __forceinline__ float waveAllSum(float v) {
#pragma unroll
  for (int o = 1; o < 32; o <<= 1) v += __shfl_xor(v, o, 32);
  return v;
}
__device__ __forceinline__ float waveRedSum(float v) {
#pragma unroll
  for (int o = 16; o > 0; o >>= 1) v += __shfl_down(v, o, 32);
  return v;
}
__device__ __forceinline__ float softplusf(float x) {
  return fmaxf(x, 0.f) + __logf(1.f + __expf(-fabsf(x)));
}

// CDNA5 async global->LDS copy (16B per lane), tracked by ASYNCcnt.
__device__ __forceinline__ void async_g2l_b128(uint32_t lds_byte, const void* g) {
  asm volatile("global_load_async_to_lds_b128 %0, %1, off"
               :: "v"(lds_byte), "v"(g) : "memory");
}
__device__ __forceinline__ void wait_asynccnt0() {
  asm volatile("s_wait_asynccnt 0x0" ::: "memory");
}

// ---------------------------------------------------------------------------
// Zero the scalar output
// ---------------------------------------------------------------------------
__global__ void k_zero(float* out, int n) {
  int t = blockIdx.x * blockDim.x + threadIdx.x;
  if (t < n) out[t] = 0.f;
}

// ---------------------------------------------------------------------------
// Per-layer constants: softmax(p_k_raw), c0 = p0 + eps*(1-p0).
// consts[l*16 + 0] = c0 ; consts[l*16 + k] = p_k (k=1..7)
// ---------------------------------------------------------------------------
__global__ void k_consts(const float* __restrict__ pkr, float* __restrict__ consts, int L) {
  if (threadIdx.x != 0 || blockIdx.x != 0) return;
  for (int l = 0; l < L; ++l) {
    const float* p = pkr + l * 8;
    float mx = p[0];
    for (int t = 1; t < 8; ++t) mx = fmaxf(mx, p[t]);
    float e[8], s = 0.f;
    for (int t = 0; t < 8; ++t) { e[t] = __expf(p[t] - mx); s += e[t]; }
    float inv = 1.f / s;
    float p0 = e[0] * inv;
    consts[l * 16 + 0] = p0 + EPSV * (1.f - p0);
    for (int t = 1; t < 8; ++t) consts[l * 16 + t] = e[t] * inv;
  }
}

// ---------------------------------------------------------------------------
// Row softmax over K=128 (one wave per row, 4 elems/lane).
// hier==1: also fold the dyadic pooling pyramid weighted by p_k
//          out[j] = p7*z_j + sum_{k=1..6} p_k * blocksum_{2^k}(z, j)
// hier==0: out = softmax only.  Output f16.
// ---------------------------------------------------------------------------
__global__ __launch_bounds__(256) void k_rows(const float* __restrict__ in,
                                              const float* __restrict__ consts,
                                              _Float16* __restrict__ out,
                                              int totalRows, int rowsPerLayer, int hier) {
  int w = (blockIdx.x * blockDim.x + threadIdx.x) >> 5;
  int lane = threadIdx.x & 31;
  if (w >= totalRows) return;
  int layer = w / rowsPerLayer;

  const float* row = in + (size_t)w * 128 + lane * 4;
  v4f x = *(const v4f*)row;

  float mx = fmaxf(fmaxf(x[0], x[1]), fmaxf(x[2], x[3]));
  mx = waveAllMax(mx);
  float e0 = __expf(x[0] - mx), e1 = __expf(x[1] - mx),
        e2 = __expf(x[2] - mx), e3 = __expf(x[3] - mx);
  float s = waveAllSum(e0 + e1 + e2 + e3);
  float inv = 1.f / s;
  float z0 = e0 * inv, z1 = e1 * inv, z2 = e2 * inv, z3 = e3 * inv;

  float o0, o1, o2, o3;
  if (hier) {
    float s1a = z0 + z1, s1b = z2 + z3;       // block-of-2 sums
    float s2  = s1a + s1b;                    // block-of-4 (exactly this lane)
    float b8  = s2  + __shfl_xor(s2,  1, 32); // block-of-8
    float b16 = b8  + __shfl_xor(b8,  2, 32); // block-of-16
    float b32 = b16 + __shfl_xor(b16, 4, 32); // block-of-32
    float b64 = b32 + __shfl_xor(b32, 8, 32); // block-of-64
    const float* c = consts + layer * 16;
    float p1 = c[1], p2 = c[2], p3 = c[3], p4 = c[4], p5 = c[5], p6 = c[6], p7 = c[7];
    float common = p2 * s2 + p3 * b8 + p4 * b16 + p5 * b32 + p6 * b64;
    o0 = p7 * z0 + p1 * s1a + common;
    o1 = p7 * z1 + p1 * s1a + common;
    o2 = p7 * z2 + p1 * s1b + common;
    o3 = p7 * z3 + p1 * s1b + common;
  } else {
    o0 = z0; o1 = z1; o2 = z2; o3 = z3;
  }
  v4h o; o[0] = (_Float16)o0; o[1] = (_Float16)o1; o[2] = (_Float16)o2; o[3] = (_Float16)o3;
  *(v4h*)(out + (size_t)w * 128 + lane * 4) = o;
}

// ---------------------------------------------------------------------------
// a[l*N1+i] = softmax(latent_z1[i,:])[l] * softplus(L_raw)
// b[l*N2+j] = softmax(latent_w1[j,:])[l] + eps
// ---------------------------------------------------------------------------
__global__ __launch_bounds__(256) void k_ab(const float* __restrict__ lz,
                                            const float* __restrict__ lw,
                                            const float* __restrict__ Lraw,
                                            float* __restrict__ avec,
                                            float* __restrict__ bvec,
                                            int N1, int N2, int L) {
  int i = blockIdx.x * blockDim.x + threadIdx.x;
  float Ls = fmaxf(Lraw[0], 0.f) + log1pf(__expf(-fabsf(Lraw[0])));
  float ev[8];
  if (i < N1) {
    float mx = lz[(size_t)i * L];
    for (int l = 1; l < L; ++l) mx = fmaxf(mx, lz[(size_t)i * L + l]);
    float s = 0.f;
    for (int l = 0; l < L; ++l) { ev[l] = __expf(lz[(size_t)i * L + l] - mx); s += ev[l]; }
    float inv = 1.f / s;
    for (int l = 0; l < L; ++l) avec[(size_t)l * N1 + i] = ev[l] * inv * Ls;
  }
  if (i < N2) {
    float mx = lw[(size_t)i * L];
    for (int l = 1; l < L; ++l) mx = fmaxf(mx, lw[(size_t)i * L + l]);
    float s = 0.f;
    for (int l = 0; l < L; ++l) { ev[l] = __expf(lw[(size_t)i * L + l] - mx); s += ev[l]; }
    float inv = 1.f / s;
    for (int l = 0; l < L; ++l) bvec[(size_t)l * N2 + i] = ev[l] * inv + EPSV;
  }
}

// ---------------------------------------------------------------------------
// Fused main kernel: per block a 256(M) x 128(N) tile of M = c0 + Z' W^T.
// 8 waves; wave w owns 32 rows = 2 row-groups of 16, all 128 cols.
//
// B tile (shared by all waves) staged once into LDS with
// global_load_async_to_lds_b128 (ASYNCcnt); A fragments preloaded from global
// while the copy is in flight.  Per K-step all 8 B fragments are loaded into
// registers first, then 16 WMMAs issue back-to-back (each B frag feeds both
// row-groups) -> one DScnt wait per 16 WMMAs, and 2x reuse per LDS byte.
// LDS row stride = 136 halfs (272B = 17 x 16B) -> conflict-free ds_load_b128.
//
// Fragment layouts per CDNA5 ISA 7.12.2 (wave32):
//   A 16x32 f16 : lane -> M = lane%16; elem e -> K = e%8 + (e/8)*16 + (lane/16)*8
//   B 32x16 f16 : lane -> N = lane%16; elem e -> K = e%8 + (e/8)*16 + (lane/16)*8
//   C 16x16 f32 : lane -> N = lane%16; vgpr r -> M = r + (lane/16)*8
// ---------------------------------------------------------------------------
#define LDS_STRIDE 136  // halfs; 272B = 17*16B, row advances 4 banks

__global__ __launch_bounds__(256) void k_main(const _Float16* __restrict__ Zp,
                                              const _Float16* __restrict__ Wh,
                                              const float* __restrict__ avec,
                                              const float* __restrict__ bvec,
                                              const float* __restrict__ gamma,
                                              const float* __restrict__ delta,
                                              const float* __restrict__ consts,
                                              float* __restrict__ out,
                                              int N1, int N2, int L) {
  __shared__ __align__(16) _Float16 tB[128 * LDS_STRIDE];

  const int K = 128;
  const int layer = blockIdx.z;
  const int m0 = blockIdx.x * 256;
  const int n0 = blockIdx.y * 128;
  const int tid  = threadIdx.x;
  const int wave = tid >> 5;
  const int lane = tid & 31;
  const int hlf  = lane >> 4;
  const int l16  = lane & 15;
  const int mw0 = m0 + wave * 32;          // first of the wave's 2 row-groups

  const _Float16* Zl = Zp + (size_t)layer * N1 * K;
  const _Float16* Wl = Wh + (size_t)layer * N2 * K;

  // ---- stage B tile: 128 rows x 256B = 2048 x 16B chunks, 8 per thread ----
  {
    const _Float16* src = Wl + (size_t)n0 * K;
#pragma unroll
    for (int c = 0; c < 8; ++c) {
      const int q   = c * 256 + tid;       // chunk id; 16 lanes span one row
      const int row = q >> 4;
      const int col = q & 15;
      const _Float16* g = src + (size_t)row * K + col * 8;
      const uint32_t  l = (uint32_t)(uintptr_t)(tB + row * LDS_STRIDE + col * 8);
      async_g2l_b128(l, g);
    }
  }

  // ---- preload all A fragments (2 row-groups x 4 K-steps), overlaps staging ----
  Frag16 aF[2][4];
#pragma unroll
  for (int g = 0; g < 2; ++g) {
    const _Float16* arow = Zl + (size_t)(mw0 + g * 16 + l16) * K;
#pragma unroll
    for (int kk = 0; kk < 4; ++kk) {
      const int K0 = kk * 32;
      aF[g][kk].h[0] = *(const v8h*)(arow + K0 + hlf * 8);
      aF[g][kk].h[1] = *(const v8h*)(arow + K0 + 16 + hlf * 8);
    }
  }

  wait_asynccnt0();
  __syncthreads();

  // ---- GEMM: 4 K-steps; per step load 8 B frags, then 16 WMMAs ----
  v8f acc[2][8];
#pragma unroll
  for (int g = 0; g < 2; ++g)
#pragma unroll
    for (int f = 0; f < 8; ++f) acc[g][f] = {};

#pragma unroll
  for (int kk = 0; kk < 4; ++kk) {
    const int K0 = kk * 32;
    Frag16 b[8];
#pragma unroll
    for (int f = 0; f < 8; ++f) {
      const _Float16* bp = tB + (f * 16 + l16) * LDS_STRIDE + K0 + hlf * 8;
      b[f].h[0] = *(const v8h*)(bp);
      b[f].h[1] = *(const v8h*)(bp + 16);
    }
#pragma unroll
    for (int g = 0; g < 2; ++g)
#pragma unroll
      for (int f = 0; f < 8; ++f)
        acc[g][f] = __builtin_amdgcn_wmma_f32_16x16x32_f16(
            false, aF[g][kk].v, false, b[f].v, (short)0, acc[g][f], false, false);
  }

  // ---- epilogue: v = a_i*b_j*(c0+acc) + g_i + d_j ; sum softplus, skip diag ----
  const float c0 = consts[layer * 16 + 0];
  float lsum = 0.f;
#pragma unroll
  for (int g = 0; g < 2; ++g) {
    float ai8[8], gi8[8];
#pragma unroll
    for (int r = 0; r < 8; ++r) {
      const int i = mw0 + g * 16 + hlf * 8 + r;
      ai8[r] = avec[(size_t)layer * N1 + i];
      gi8[r] = gamma[(size_t)i * L + layer];
    }
#pragma unroll
    for (int f = 0; f < 8; ++f) {
      const int j = n0 + f * 16 + l16;
      const float bj = bvec[(size_t)layer * N2 + j];
      const float dj = delta[(size_t)j * L + layer];
#pragma unroll
      for (int r = 0; r < 8; ++r) {
        const int i = mw0 + g * 16 + hlf * 8 + r;
        const float m = c0 + acc[g][f][r];
        const float v = fmaf(ai8[r] * bj, m, gi8[r] + dj);
        if (!(N1 == N2 && i == j)) lsum += softplusf(v);
      }
    }
  }
  lsum = waveRedSum(lsum);
  if (lane == 0) atomicAdd(out, -lsum);  // contributes -z_pdist1
}

// ---------------------------------------------------------------------------
// Sparse term: one wave per (layer, entry).
// val = a_i*b_j*(c0 + dot128(Z'[i], W[j])) + g_i + d_j   (accumulate +)
// ---------------------------------------------------------------------------
__global__ __launch_bounds__(256) void k_sparse(const _Float16* __restrict__ Zp,
                                                const _Float16* __restrict__ Wh,
                                                const float* __restrict__ avec,
                                                const float* __restrict__ bvec,
                                                const float* __restrict__ gamma,
                                                const float* __restrict__ delta,
                                                const float* __restrict__ consts,
                                                const int* __restrict__ si,
                                                const int* __restrict__ sj,
                                                float* __restrict__ out,
                                                int N1, int N2, int L, int E) {
  const int w = (blockIdx.x * blockDim.x + threadIdx.x) >> 5;
  const int lane = threadIdx.x & 31;
  const long long total = (long long)L * E;
  if (w >= total) return;
  const int layer = w / E;
  const int e = w - layer * E;
  const int i = si[(size_t)layer * E + e];
  const int j = sj[(size_t)layer * E + e];

  const _Float16* zr = Zp + ((size_t)layer * N1 + i) * 128 + lane * 4;
  const _Float16* wr = Wh + ((size_t)layer * N2 + j) * 128 + lane * 4;
  v4h za = *(const v4h*)zr;
  v4h wa = *(const v4h*)wr;
  float dot = (float)za[0] * (float)wa[0] + (float)za[1] * (float)wa[1] +
              (float)za[2] * (float)wa[2] + (float)za[3] * (float)wa[3];
  dot = waveRedSum(dot);
  if (lane == 0) {
    const float c0 = consts[layer * 16 + 0];
    const float val = avec[(size_t)layer * N1 + i] * bvec[(size_t)layer * N2 + j] * (c0 + dot) +
                      gamma[(size_t)i * L + layer] + delta[(size_t)j * L + layer];
    atomicAdd(out, val);  // contributes +z_pdist2
  }
}

// ---------------------------------------------------------------------------
// Launcher
// ---------------------------------------------------------------------------
extern "C" void kernel_launch(void* const* d_in, const int* in_sizes, int n_in,
                              void* d_out, int out_size, void* d_ws, size_t ws_size,
                              hipStream_t stream) {
  const float* us    = (const float*)d_in[0];
  const float* vs    = (const float*)d_in[1];
  const float* gamma = (const float*)d_in[2];
  const float* delta = (const float*)d_in[3];
  const float* lz    = (const float*)d_in[4];
  const float* lw    = (const float*)d_in[5];
  const float* pkr   = (const float*)d_in[6];
  const float* Lraw  = (const float*)d_in[7];
  const int*   si    = (const int*)d_in[8];
  const int*   sj    = (const int*)d_in[9];
  float* out = (float*)d_out;

  const int L  = in_sizes[6] / 8;          // p_k_raw is (L, 8)
  const int N1 = in_sizes[2] / L;          // gamma (N1, L)
  const int N2 = in_sizes[3] / L;          // delta (N2, L)
  const int E  = in_sizes[8] / L;          // sparse_is (L, E)
  const int K  = 128;                      // 2^TH

  // Workspace carve-up (256B aligned)
  uintptr_t p = ((uintptr_t)d_ws + 255) & ~(uintptr_t)255;
  float* consts = (float*)p;  p = (p + (size_t)L * 16 * 4 + 255) & ~(uintptr_t)255;
  float* avec   = (float*)p;  p = (p + (size_t)L * N1 * 4 + 255) & ~(uintptr_t)255;
  float* bvec   = (float*)p;  p = (p + (size_t)L * N2 * 4 + 255) & ~(uintptr_t)255;
  _Float16* Zp  = (_Float16*)p; p = (p + (size_t)L * N1 * K * 2 + 255) & ~(uintptr_t)255;
  _Float16* Wh  = (_Float16*)p;
  (void)ws_size; (void)n_in;

  k_zero<<<1, 64, 0, stream>>>(out, out_size);
  k_consts<<<1, 1, 0, stream>>>(pkr, consts, L);

  const int rowsZ = L * N1, rowsW = L * N2;
  k_rows<<<(rowsZ + 7) / 8, 256, 0, stream>>>(us, consts, Zp, rowsZ, N1, 1);
  k_rows<<<(rowsW + 7) / 8, 256, 0, stream>>>(vs, consts, Wh, rowsW, N2, 0);

  const int Nmax = N1 > N2 ? N1 : N2;
  k_ab<<<(Nmax + 255) / 256, 256, 0, stream>>>(lz, lw, Lraw, avec, bvec, N1, N2, L);

  dim3 grid(N1 / 256, N2 / 128, L);
  k_main<<<grid, 256, 0, stream>>>(Zp, Wh, avec, bvec, gamma, delta, consts, out, N1, N2, L);

  const long long tw = (long long)L * E;
  k_sparse<<<(unsigned)((tw + 7) / 8), 256, 0, stream>>>(Zp, Wh, avec, bvec, gamma, delta,
                                                         consts, si, sj, out, N1, N2, L, E);
}
